// Attention_2808908611625
// MI455X (gfx1250) — matrix operand order repaired and checked by reading, not compiled, not run on hardware
//
#include <hip/hip_runtime.h>

// ---------------------------------------------------------------------------
// MI455X (gfx1250) attention:  out = softmax(QK^T/sqrt(d) + mask*-1e9) @ V
// All matmuls via v_wmma_f32_16x16x32_bf16 (wave32, f32 accumulate).
// Double-buffered LDS tiles, ds_load_b128 fragment loads (16B-aligned,
// bank-conflict-free padded stride of 40 halves).
// ---------------------------------------------------------------------------

typedef __attribute__((ext_vector_type(16))) __bf16 v16bf;
typedef __attribute__((ext_vector_type(8)))  float  v8f;

#define BM 128
#define BN 128
#define BK 32
#define LDS_STRIDE 40   // 80 B rows: 16B-aligned for b128; 20*r mod 64 hits
                        // distinct multiples of 4 for r=0..15 -> conflict-free

__device__ __forceinline__ unsigned short f2bf(float f) {
  union { float f; unsigned int u; } c; c.f = f;
  unsigned int u = c.u;
  u += 0x7FFFu + ((u >> 16) & 1u);          // round-to-nearest-even
  return (unsigned short)(u >> 16);
}

// A fragment, 16x32 bf16 (ISA 7.12.2): lanes 0-15 row=lane, K in {0..7,16..23};
// lanes 16-31 row=lane-16, K in {8..15,24..31}. Both 8-half runs contiguous
// and 16B-aligned -> two ds_load_b128.
__device__ __forceinline__ v16bf load_frag_a(const unsigned short* s, int baseRow, int lane) {
  const int row = baseRow + (lane & 15);
  const int kb  = (lane < 16) ? 0 : 8;
  union { uint4 q[2]; v16bf v; } r;
  const unsigned short* p = s + row * LDS_STRIDE + kb;
  r.q[0] = *(const uint4*)(p);        // halves kb .. kb+7    (K 0..7  / 8..15)
  r.q[1] = *(const uint4*)(p + 16);   // halves kb+16..kb+23  (K 16..23/24..31)
  return r.v;
}

// B fragment, 32x16 bf16: lane = column; lanes 0-15 hold K={0..7,16..23}, lanes 16-31
// hold K={8..15,24..31} (matches A). LDS holds B as [n][k] -> two b128.
__device__ __forceinline__ v16bf load_frag_b(const unsigned short* s, int baseCol, int lane) {
  const int n  = baseCol + (lane & 15);
  const int kb = (lane < 16) ? 0 : 8;
  union { uint4 q[2]; v16bf v; } r;
  const unsigned short* p = s + n * LDS_STRIDE + kb;
  r.q[0] = *(const uint4*)(p);
  r.q[1] = *(const uint4*)(p + 16);
  return r.v;
}

// ---------------------------------------------------------------------------
// Stage one BMxBK A-tile and BKxBN (or BNxBK) B-tile into LDS as bf16.
// ---------------------------------------------------------------------------
template<bool A_F32, bool B_KN, bool B_F32>
__device__ __forceinline__ void stage_tiles(
    unsigned short* __restrict__ sA, unsigned short* __restrict__ sB,
    const char* __restrict__ Ab, const char* __restrict__ Bb,
    int m0, int n0, int kb, int N, int Kd, int tid)
{
  // ---- A tile (BM x BK), row-major copy, ds_store_b64 ----
  for (int i = tid; i < BM * (BK / 4); i += 256) {
    const int r  = i >> 3;
    const int kc = (i & 7) << 2;
    uint2 w;
    if (A_F32) {
      const float4 f = *(const float4*)(Ab + ((size_t)(m0 + r) * Kd + kb + kc) * 4);
      w.x = (unsigned int)f2bf(f.x) | ((unsigned int)f2bf(f.y) << 16);
      w.y = (unsigned int)f2bf(f.z) | ((unsigned int)f2bf(f.w) << 16);
    } else {
      w = *(const uint2*)(Ab + ((size_t)(m0 + r) * Kd + kb + kc) * 2);
    }
    *(uint2*)&sA[r * LDS_STRIDE + kc] = w;
  }
  // ---- B tile into LDS as [n][k] ----
  if (B_KN) {  // B is [K,N] row-major: coalesced read along n, transpose-scatter
    for (int i = tid; i < BK * (BN / 4); i += 256) {
      const int kr = i >> 5;
      const int nc = (i & 31) << 2;
      if (B_F32) {
        const float4 f = *(const float4*)(Bb + ((size_t)(kb + kr) * N + n0 + nc) * 4);
        sB[(nc + 0) * LDS_STRIDE + kr] = f2bf(f.x);
        sB[(nc + 1) * LDS_STRIDE + kr] = f2bf(f.y);
        sB[(nc + 2) * LDS_STRIDE + kr] = f2bf(f.z);
        sB[(nc + 3) * LDS_STRIDE + kr] = f2bf(f.w);
      } else {
        const uint2 u = *(const uint2*)(Bb + ((size_t)(kb + kr) * N + n0 + nc) * 2);
        sB[(nc + 0) * LDS_STRIDE + kr] = (unsigned short)(u.x);
        sB[(nc + 1) * LDS_STRIDE + kr] = (unsigned short)(u.x >> 16);
        sB[(nc + 2) * LDS_STRIDE + kr] = (unsigned short)(u.y);
        sB[(nc + 3) * LDS_STRIDE + kr] = (unsigned short)(u.y >> 16);
      }
    }
  } else {     // B is [N,K] row-major (K-matrix for scores): straight copy
    for (int i = tid; i < BN * (BK / 4); i += 256) {
      const int n  = i >> 3;
      const int kc = (i & 7) << 2;
      const uint2 w = *(const uint2*)(Bb + ((size_t)(n0 + n) * Kd + kb + kc) * 2);
      *(uint2*)&sB[n * LDS_STRIDE + kc] = w;
    }
  }
}

// ---------------------------------------------------------------------------
// Tiled GEMM:  C[M,N] = epilogue( A[M,K] * B )
//   EPI 0 : +bias[col], store bf16        (Q/K/V projections)
//   EPI 1 : *scale + mask*-1e9, store f32 (attention scores)
//   EPI 2 : store f32                     (P @ V output)
// blockDim = 256 (8 waves, wave32); wave grid 4(M) x 2(N); wave tile 32x64.
// ---------------------------------------------------------------------------
template<bool A_F32, bool B_KN, bool B_F32, int EPI>
__global__ __launch_bounds__(256) void gemm_wmma_kernel(
    const void* __restrict__ Ap_, const void* __restrict__ Bp_,
    const float* __restrict__ bias, const float* __restrict__ maskp_,
    void* __restrict__ Cp_, int M, int N, int Kd, float scale,
    long long strA, long long strB, long long strC, long long strM)
{
  __shared__ alignas(16) unsigned short sA[2][BM * LDS_STRIDE];
  __shared__ alignas(16) unsigned short sB[2][BN * LDS_STRIDE];

  const int tid  = threadIdx.x;
  const int lane = tid & 31;
  const int wave = tid >> 5;
  const int wm   = wave >> 1;                 // 0..3
  const int wn   = wave & 1;                  // 0..1
  const int m0   = blockIdx.y * BM;
  const int n0   = blockIdx.x * BN;
  const int z    = blockIdx.z;

  const char*  Ab    = (const char*)Ap_ + (size_t)z * strA * (A_F32 ? 4 : 2);
  const char*  Bb    = (const char*)Bp_ + (size_t)z * strB * (B_F32 ? 4 : 2);
  const float* maskp = maskp_ ? (maskp_ + (size_t)z * strM) : nullptr;
  char*        Cb    = (char*)Cp_ + (size_t)z * strC * (EPI == 0 ? 2 : 4);

  v8f acc[2][4];
  const v8f vzero = {0.f, 0.f, 0.f, 0.f, 0.f, 0.f, 0.f, 0.f};
#pragma unroll
  for (int s2 = 0; s2 < 2; ++s2)
#pragma unroll
    for (int t = 0; t < 4; ++t) acc[s2][t] = vzero;

  // prologue: stage K-tile 0 into buffer 0
  stage_tiles<A_F32, B_KN, B_F32>(sA[0], sB[0], Ab, Bb, m0, n0, 0, N, Kd, tid);
  __syncthreads();

  int buf = 0;
  for (int kb = 0; kb < Kd; kb += BK) {
    // stage next tile into the other buffer while we compute this one
    if (kb + BK < Kd) {
      stage_tiles<A_F32, B_KN, B_F32>(sA[buf ^ 1], sB[buf ^ 1], Ab, Bb,
                                      m0, n0, kb + BK, N, Kd, tid);
      if (kb + 2 * BK < Kd)   // global_prefetch_b8 of the tile after that
        __builtin_prefetch(Ab + ((size_t)(m0 + (tid & 127)) * Kd + kb + 2 * BK) *
                                   (A_F32 ? 4 : 2), 0, 1);
    }

    const v16bf a0 = load_frag_a(sA[buf], wm * 32,      lane);
    const v16bf a1 = load_frag_a(sA[buf], wm * 32 + 16, lane);
#pragma unroll
    for (int t = 0; t < 4; ++t) {
      const v16bf bt = load_frag_b(sB[buf], wn * 64 + t * 16, lane);
      acc[0][t] = __builtin_amdgcn_wmma_f32_16x16x32_bf16(
          false, a0, false, bt, (short)0, acc[0][t], false, false);
      acc[1][t] = __builtin_amdgcn_wmma_f32_16x16x32_bf16(
          false, a1, false, bt, (short)0, acc[1][t], false, false);
    }
    __syncthreads();   // next buffer fully staged; this buffer free to overwrite
    buf ^= 1;
  }

  // ---- epilogue: C/D layout = VGPR r -> row r (+8 for lanes 16-31), col = lane&15
  const int rowW = m0 + wm * 32 + ((lane < 16) ? 0 : 8);
  const int colW = n0 + wn * 64 + (lane & 15);
#pragma unroll
  for (int s2 = 0; s2 < 2; ++s2)
#pragma unroll
    for (int t = 0; t < 4; ++t)
#pragma unroll
      for (int r = 0; r < 8; ++r) {
        const int row = rowW + s2 * 16 + r;
        const int col = colW + t * 16;
        const size_t idx = (size_t)row * N + col;
        float v = acc[s2][t][r];
        if (EPI == 0) {
          v += bias[col];
          ((unsigned short*)Cb)[idx] = f2bf(v);
        } else if (EPI == 1) {
          v = v * scale + maskp[idx] * -1e9f;
          ((float*)Cb)[idx] = v;
        } else {
          ((float*)Cb)[idx] = v;
        }
      }
}

// ---------------------------------------------------------------------------
// Row softmax: one 256-thread block per row of 2048 f32 scores -> bf16 probs.
// ---------------------------------------------------------------------------
__global__ __launch_bounds__(256) void softmax_kernel(const float* __restrict__ S,
                                                      unsigned short* __restrict__ P) {
  __shared__ float red[8];
  const int tid = threadIdx.x;
  const size_t base = (size_t)blockIdx.x * 2048;
  float v[8];
  float m = -3.4e38f;
#pragma unroll
  for (int i = 0; i < 8; ++i) { v[i] = S[base + tid + i * 256]; m = fmaxf(m, v[i]); }
#pragma unroll
  for (int off = 16; off > 0; off >>= 1) m = fmaxf(m, __shfl_xor(m, off, 32));
  if ((tid & 31) == 0) red[tid >> 5] = m;
  __syncthreads();
  float rowmax = red[0];
#pragma unroll
  for (int i = 1; i < 8; ++i) rowmax = fmaxf(rowmax, red[i]);
  __syncthreads();
  float sum = 0.f;
#pragma unroll
  for (int i = 0; i < 8; ++i) { v[i] = __expf(v[i] - rowmax); sum += v[i]; }
#pragma unroll
  for (int off = 16; off > 0; off >>= 1) sum += __shfl_xor(sum, off, 32);
  if ((tid & 31) == 0) red[tid >> 5] = sum;
  __syncthreads();
  float tot = 0.f;
#pragma unroll
  for (int i = 0; i < 8; ++i) tot += red[i];
  const float inv = 1.0f / tot;
#pragma unroll
  for (int i = 0; i < 8; ++i) P[base + tid + i * 256] = f2bf(v[i] * inv);
}

// ---------------------------------------------------------------------------
extern "C" void kernel_launch(void* const* d_in, const int* in_sizes, int n_in,
                              void* d_out, int out_size, void* d_ws, size_t ws_size,
                              hipStream_t stream) {
  (void)in_sizes; (void)n_in; (void)out_size; (void)ws_size;
  const float* primary = (const float*)d_in[0];
  const float* context = (const float*)d_in[1];
  const float* maskp   = (const float*)d_in[2];
  const float* Wq = (const float*)d_in[3];
  const float* bq = (const float*)d_in[4];
  const float* Wk = (const float*)d_in[5];
  const float* bk = (const float*)d_in[6];
  const float* Wv = (const float*)d_in[7];
  const float* bv = (const float*)d_in[8];

  const int Bn = 4, LQ = 2048, LKV = 2048, D = 1024;
  const size_t nQ = (size_t)Bn * LQ * D;     // 8M bf16 elems per tensor
  const size_t nS = (size_t)Bn * LQ * LKV;   // 16M score elems

  // workspace: Q,K,V bf16 (48 MB) + S f32 (64 MB) + P bf16 (32 MB) = 144 MB
  unsigned short* Qb = (unsigned short*)d_ws;
  unsigned short* Kb = Qb + nQ;
  unsigned short* Vb = Kb + nQ;
  float*          Sb = (float*)(Vb + nQ);
  unsigned short* Pb = (unsigned short*)(Sb + nS);

  const dim3 blk(256);

  // Q/K/V projections (batch folded into M = B*L = 8192)
  gemm_wmma_kernel<true, true, true, 0><<<dim3(D / BN, (Bn * LQ) / BM, 1), blk, 0, stream>>>(
      primary, Wq, bq, nullptr, Qb, Bn * LQ, D, D, 1.0f, 0, 0, 0, 0);
  gemm_wmma_kernel<true, true, true, 0><<<dim3(D / BN, (Bn * LKV) / BM, 1), blk, 0, stream>>>(
      context, Wk, bk, nullptr, Kb, Bn * LKV, D, D, 1.0f, 0, 0, 0, 0);
  gemm_wmma_kernel<true, true, true, 0><<<dim3(D / BN, (Bn * LKV) / BM, 1), blk, 0, stream>>>(
      context, Wv, bv, nullptr, Vb, Bn * LKV, D, D, 1.0f, 0, 0, 0, 0);

  // S = Q K^T / sqrt(D) + mask * -1e9   (B = K-matrix is [N,K] row-major)
  gemm_wmma_kernel<false, false, false, 1><<<dim3(LKV / BN, LQ / BM, Bn), blk, 0, stream>>>(
      Qb, Kb, nullptr, maskp, Sb, LQ, LKV, D, 0.03125f,
      (long long)LQ * D, (long long)LKV * D, (long long)LQ * LKV, (long long)LQ * LKV);

  // P = softmax(S)  (bf16)
  softmax_kernel<<<dim3(Bn * LQ), blk, 0, stream>>>(Sb, Pb);

  // out = P V   (B = V is [K,N] row-major, staged transposed)
  gemm_wmma_kernel<false, true, false, 2><<<dim3(D / BN, LQ / BM, Bn), blk, 0, stream>>>(
      Pb, Vb, nullptr, nullptr, (float*)d_out, LQ, D, LKV, 1.0f,
      (long long)LQ * LKV, (long long)LKV * D, (long long)LQ * D, 0);
}
